// MoELayer_43963285242291
// MI455X (gfx1250) — matrix operand order repaired and checked by reading, not compiled, not run on hardware
//
#include <hip/hip_runtime.h>
#include <hip/hip_bf16.h>

// ---------------------------------------------------------------------------
// MoE (E=8, top-2, DIM=1024, HID=2048, 8192 tokens) for MI455X / gfx1250.
// Gathered grouped GEMMs in bf16 WMMA (v_wmma_f32_16x16x32_bf16), fp32 accum.
//  - 206 GFLOP sparse; bf16 weights (96MB) fit in the 192MB L2.
//  - A tiles staged global->LDS via GLOBAL_LOAD_ASYNC_TO_LDS_B128 (ASYNCcnt).
//  - B tiles transposed in registers, stored with ds_store_b128 (not b16).
//  - Deterministic: no fp atomics; per-entry y + per-token combine.
// ---------------------------------------------------------------------------

#define E_    8
#define TOPK  2
#define DIM_  1024
#define HID_  2048
#define NTOK  (4 * 2048)          // B*T = 8192
#define NENT  (NTOK * TOPK)       // 16384 routed entries

typedef unsigned short u16;
typedef __attribute__((ext_vector_type(4)))  unsigned int u32x4;  // 16B unit
typedef __attribute__((ext_vector_type(4)))  float  f32x4;
typedef __attribute__((ext_vector_type(4)))  u16    u16x4;
typedef __attribute__((ext_vector_type(16))) __bf16 v16bf;
typedef __attribute__((ext_vector_type(8)))  __bf16 v8bf;
typedef __attribute__((ext_vector_type(8)))  float  v8f;

// GEMM tiling
#define TM   128      // token-tile rows
#define TK   32       // K step (one WMMA depth)
#define TN   128      // output-column tile
#define ASTR 40       // LDS row stride (elems), padded, keeps 16B alignment
#define BSTR 40

__device__ __forceinline__ u16 f2bf(float f) {
  unsigned u = __builtin_bit_cast(unsigned, f);
  u += 0x7FFFu + ((u >> 16) & 1u);            // round-to-nearest-even
  return (u16)(u >> 16);
}

__device__ __forceinline__ v8f wmma_bf16(v16bf a, v16bf b, v8f c) {
  // D = A(16x32 bf16) * B(32x16 bf16) + C(16x16 f32)
  return __builtin_amdgcn_wmma_f32_16x16x32_bf16(
      /*neg_a=*/false, a, /*neg_b=*/false, b,
      /*c_mod=*/(short)0, c, /*reuse_a=*/false, /*reuse_b=*/false);
}

// Async copy 32B global -> LDS (per lane). INST_OFFSET applies to both the
// LDS and the memory address (ISA 10.7 async pseudocode), so one offset:16
// shifts both sides. Tracked by ASYNCcnt.
__device__ __forceinline__ void async_copy32(unsigned ldsOff, const u16* gptr) {
  asm volatile(
      "global_load_async_to_lds_b128 %0, %1, off\n\t"
      "global_load_async_to_lds_b128 %0, %1, off offset:16"
      :: "v"(ldsOff), "v"(gptr) : "memory");
}
__device__ __forceinline__ void wait_async0() {
  asm volatile("s_wait_asynccnt 0x0" ::: "memory");
}

// A fragment: sA is [rows][ASTR] row-major (K contiguous).
// ISA 16-bit A 16x32 layout: lane<16 -> K 0..7 & 16..23 ; lane>=16 -> +8.
__device__ __forceinline__ v16bf ldfragA(const u16* sA, int row, int lane) {
  const u16* p = sA + row * ASTR + ((lane >> 4) << 3);
  union { v16bf v; v8bf h[2]; } u;
  u.h[0] = *(const v8bf*)(p);
  u.h[1] = *(const v8bf*)(p + 16);
  return u.v;
}

// B fragment: sB holds B^T, i.e. [n][k] (K contiguous per column of B).
// Dense B 32x16: lane<16 -> N=lane, K 0..7 & 16..23 ; lane>=16 -> N=lane-16, +8.
__device__ __forceinline__ v16bf ldfragB(const u16* sB, int col, int lane) {
  const u16* p = sB + col * BSTR + ((lane >> 4) << 3);
  union { v16bf v; v8bf h[2]; } u;
  u.h[0] = *(const v8bf*)(p);
  u.h[1] = *(const v8bf*)(p + 16);
  return u.v;
}

// ---------------------------------------------------------------------------
__global__ void zero_meta(int* counts, int* fill) {
  int t = threadIdx.x;
  if (t < E_) { counts[t] = 0; fill[t] = 0; }
}

// Wave-per-token router: logits = x @ rw^T + rb, top-2, softmax.
__global__ __launch_bounds__(256)
void router_kernel(const float* __restrict__ x, const float* __restrict__ rw,
                   const float* __restrict__ rb, int* __restrict__ tokExpert,
                   float* __restrict__ tokWeight, int* __restrict__ counts) {
  int gt   = blockIdx.x * (blockDim.x >> 5) + (threadIdx.x >> 5);
  int lane = threadIdx.x & 31;
  if (gt >= NTOK) return;
  const float* xr = x + (size_t)gt * DIM_;
  float acc[E_];
#pragma unroll
  for (int e = 0; e < E_; ++e) acc[e] = 0.f;
  for (int d = lane; d < DIM_; d += 32) {
    float xv = xr[d];
#pragma unroll
    for (int e = 0; e < E_; ++e) acc[e] += xv * rw[e * DIM_ + d];
  }
#pragma unroll
  for (int e = 0; e < E_; ++e) {
#pragma unroll
    for (int off = 16; off > 0; off >>= 1)
      acc[e] += __shfl_xor(acc[e], off, 32);
  }
  if (lane == 0) {
    float v0 = -1e30f, v1 = -1e30f; int e0 = 0, e1 = 0;
#pragma unroll
    for (int e = 0; e < E_; ++e) {
      float v = acc[e] + rb[e];
      if (v > v0)      { v1 = v0; e1 = e0; v0 = v; e0 = e; }
      else if (v > v1) { v1 = v;  e1 = e; }
    }
    float w0 = 1.f / (1.f + __expf(v1 - v0));   // softmax over {v0,v1}
    float w1 = 1.f - w0;
    tokExpert[2 * gt]     = e0;  tokExpert[2 * gt + 1] = e1;
    tokWeight[2 * gt]     = w0;  tokWeight[2 * gt + 1] = w1;
    atomicAdd(&counts[e0], 1);   atomicAdd(&counts[e1], 1);
  }
}

__global__ void scan_kernel(const int* __restrict__ counts, int* __restrict__ offsets) {
  if (threadIdx.x == 0) {
    int o = 0;
    for (int e = 0; e < E_; ++e) { offsets[e] = o; o += counts[e]; }
    offsets[E_] = o;
  }
}

__global__ __launch_bounds__(256)
void scatter_kernel(const int* __restrict__ tokExpert, const int* __restrict__ offsets,
                    int* __restrict__ fill, int* __restrict__ entryToken,
                    int* __restrict__ tokenSlot) {
  int t = blockIdx.x * blockDim.x + threadIdx.x;
  if (t >= NTOK) return;
#pragma unroll
  for (int k = 0; k < TOPK; ++k) {
    int e = tokExpert[2 * t + k];
    int slot = offsets[e] + atomicAdd(&fill[e], 1);
    entryToken[slot] = t;
    tokenSlot[2 * t + k] = slot;
  }
}

// Vectorized fp32 -> bf16 cast: float4 in, 8B out. n must be a multiple of 4.
__global__ __launch_bounds__(256)
void cast_kernel(const float* __restrict__ src, u16* __restrict__ dst, int n) {
  int i = blockIdx.x * blockDim.x + threadIdx.x;
  const int stride = gridDim.x * blockDim.x;
  const int n4 = n >> 2;
  for (; i < n4; i += stride) {
    f32x4 f = *(const f32x4*)(src + 4 * (size_t)i);
    u16x4 o;
    o.x = f2bf(f.x); o.y = f2bf(f.y); o.z = f2bf(f.z); o.w = f2bf(f.w);
    *(u16x4*)(dst + 4 * (size_t)i) = o;
  }
}

// ---------------------------------------------------------------------------
// B staging helper: each thread owns an 8K x 2N patch of the 32x128 weight
// tile; 8 coalesced b32 global loads, register transpose, 2 ds_store_b128.
//   p  = tid & 63  -> n-pair (n = 2p, 2p+1)
//   oK = tid >> 6  -> k-octet (k = 8*oK .. 8*oK+7)
__device__ __forceinline__ void stageBT(u16* sB, const u16* pb, size_t rowStride,
                                        int p, int oK) {
  union { u32x4 q; u16 s[8]; } o0, o1;
#pragma unroll
  for (int j = 0; j < 8; ++j) {
    union { unsigned w; u16 h[2]; } v;
    v.w = *(const unsigned*)(pb + (size_t)j * rowStride);
    o0.s[j] = v.h[0];
    o1.s[j] = v.h[1];
  }
  *(u32x4*)&sB[(2 * p)     * BSTR + 8 * oK] = o0.q;
  *(u32x4*)&sB[(2 * p + 1) * BSTR + 8 * oK] = o1.q;
}

// ---------------------------------------------------------------------------
// GEMM1 + SwiGLU: for expert e, gathered token tile [TM] x cols [n0,n0+TN) of
// HID. Computes BOTH h1 (col n) and h2 (col n+HID) tiles so SwiGLU fuses.
__global__ __launch_bounds__(256)
void gemm1_swiglu_kernel(const u16* __restrict__ xb, const u16* __restrict__ W1b,
                         const float* __restrict__ b1, const int* __restrict__ entryToken,
                         const int* __restrict__ offsets, u16* __restrict__ act) {
  const int e = blockIdx.z;
  const int base = offsets[e];
  const int cnt  = offsets[e + 1] - base;
  const int m0 = blockIdx.x * TM;
  if (m0 >= cnt) return;
  const int n0 = blockIdx.y * TN;

  __shared__ u16 sA [TM * ASTR];
  __shared__ u16 sB1[TN * BSTR];
  __shared__ u16 sB2[TN * BSTR];

  const int tid = threadIdx.x;
  const int wave = tid >> 5, lane = tid & 31;
  const int wm = (wave & 3) * 32;     // 2 M-subtiles per wave
  const int wn = (wave >> 2) * 64;    // 4 N-subtiles per wave

  const v8f vz = {0.f,0.f,0.f,0.f,0.f,0.f,0.f,0.f};
  v8f acc1[2][4], acc2[2][4];
#pragma unroll
  for (int i = 0; i < 2; ++i)
#pragma unroll
    for (int j = 0; j < 4; ++j) { acc1[i][j] = vz; acc2[i][j] = vz; }

  // A staging: thread -> (row, 16-elem half of 32-wide K slab), async to LDS.
  const int rowA = tid >> 1;
  const int colA = (tid & 1) * 16;
  int mm = m0 + rowA; if (mm > cnt - 1) mm = cnt - 1;   // clamp ragged tile
  const int tok = entryToken[base + mm];
  const u16* aSrc = xb + (size_t)tok * DIM_ + colA;
  const unsigned ldsA = (unsigned)(size_t)&sA[rowA * ASTR + colA];

  // B staging: 8K x 2N patch per thread.
  const int pB = tid & 63;
  const int oK = tid >> 6;
  const size_t w1row = 2 * HID_;
  const u16* bSrc = W1b + (size_t)e * DIM_ * w1row + (size_t)(8 * oK) * w1row
                    + n0 + 2 * pB;

  for (int k0 = 0; k0 < DIM_; k0 += TK) {
    __syncthreads();                      // previous compute done
    async_copy32(ldsA, aSrc + k0);        // A: global -> LDS, no VGPR transit
    const u16* pb = bSrc + (size_t)k0 * w1row;
    if (k0 + TK < DIM_) __builtin_prefetch(pb + TK * w1row, 0, 1);
    stageBT(sB1, pb,        w1row, pB, oK);
    stageBT(sB2, pb + HID_, w1row, pB, oK);
    wait_async0();                        // A landed in LDS
    __syncthreads();                      // stage visible to all waves

    v16bf af[2];
#pragma unroll
    for (int ms = 0; ms < 2; ++ms)
      af[ms] = ldfragA(sA, wm + ms * 16 + (lane & 15), lane);
#pragma unroll
    for (int ns = 0; ns < 4; ++ns) {
      v16bf bf1 = ldfragB(sB1, wn + ns * 16 + (lane & 15), lane);
      v16bf bf2 = ldfragB(sB2, wn + ns * 16 + (lane & 15), lane);
#pragma unroll
      for (int ms = 0; ms < 2; ++ms) {
        acc1[ms][ns] = wmma_bf16(af[ms], bf1, acc1[ms][ns]);
        acc2[ms][ns] = wmma_bf16(af[ms], bf2, acc2[ms][ns]);
      }
    }
  }

  // Epilogue: C layout VGPR r -> row = r + 8*(lane>>4), col = lane&15
  const int cl = lane & 15, ch = lane >> 4;
#pragma unroll
  for (int ms = 0; ms < 2; ++ms) {
#pragma unroll
    for (int ns = 0; ns < 4; ++ns) {
      const int n = n0 + wn + ns * 16 + cl;
      const float bb1 = b1[e * (2 * HID_) + n];
      const float bb2 = b1[e * (2 * HID_) + HID_ + n];
#pragma unroll
      for (int r = 0; r < 8; ++r) {
        const int m = m0 + wm + ms * 16 + ch * 8 + r;
        if (m < cnt) {
          const float h1 = acc1[ms][ns][r] + bb1;
          const float h2 = acc2[ms][ns][r] + bb2;
          const float a  = (h1 / (1.f + __expf(-h1))) * h2;   // silu(h1)*h2
          act[(size_t)(base + m) * HID_ + n] = f2bf(a);
        }
      }
    }
  }
}

// ---------------------------------------------------------------------------
// GEMM2: y[entry] = act[entry] @ W2[e] + b2[e]   (per-entry, fp32 out)
__global__ __launch_bounds__(256)
void gemm2_kernel(const u16* __restrict__ act, const u16* __restrict__ W2b,
                  const float* __restrict__ b2, const int* __restrict__ offsets,
                  float* __restrict__ y) {
  const int e = blockIdx.z;
  const int base = offsets[e];
  const int cnt  = offsets[e + 1] - base;
  const int m0 = blockIdx.x * TM;
  if (m0 >= cnt) return;
  const int n0 = blockIdx.y * TN;

  __shared__ u16 sA[TM * ASTR];
  __shared__ u16 sB[TN * BSTR];

  const int tid = threadIdx.x;
  const int wave = tid >> 5, lane = tid & 31;
  const int wm = (wave & 3) * 32;
  const int wn = (wave >> 2) * 64;

  const v8f vz = {0.f,0.f,0.f,0.f,0.f,0.f,0.f,0.f};
  v8f acc[2][4];
#pragma unroll
  for (int i = 0; i < 2; ++i)
#pragma unroll
    for (int j = 0; j < 4; ++j) acc[i][j] = vz;

  const int rowA = tid >> 1;
  const int colA = (tid & 1) * 16;
  int mm = m0 + rowA; if (mm > cnt - 1) mm = cnt - 1;
  const u16* aSrc = act + (size_t)(base + mm) * HID_ + colA;
  const unsigned ldsA = (unsigned)(size_t)&sA[rowA * ASTR + colA];

  const int pB = tid & 63;
  const int oK = tid >> 6;
  const u16* bSrc = W2b + (size_t)e * HID_ * DIM_ + (size_t)(8 * oK) * DIM_
                    + n0 + 2 * pB;

  for (int k0 = 0; k0 < HID_; k0 += TK) {
    __syncthreads();
    async_copy32(ldsA, aSrc + k0);
    const u16* pb = bSrc + (size_t)k0 * DIM_;
    if (k0 + TK < HID_) __builtin_prefetch(pb + TK * DIM_, 0, 1);
    stageBT(sB, pb, DIM_, pB, oK);
    wait_async0();
    __syncthreads();

    v16bf af[2];
#pragma unroll
    for (int ms = 0; ms < 2; ++ms)
      af[ms] = ldfragA(sA, wm + ms * 16 + (lane & 15), lane);
#pragma unroll
    for (int ns = 0; ns < 4; ++ns) {
      v16bf bf = ldfragB(sB, wn + ns * 16 + (lane & 15), lane);
#pragma unroll
      for (int ms = 0; ms < 2; ++ms)
        acc[ms][ns] = wmma_bf16(af[ms], bf, acc[ms][ns]);
    }
  }

  const int cl = lane & 15, ch = lane >> 4;
#pragma unroll
  for (int ms = 0; ms < 2; ++ms) {
#pragma unroll
    for (int ns = 0; ns < 4; ++ns) {
      const int n = n0 + wn + ns * 16 + cl;
      const float bb = b2[e * DIM_ + n];
#pragma unroll
      for (int r = 0; r < 8; ++r) {
        const int m = m0 + wm + ms * 16 + ch * 8 + r;
        if (m < cnt) y[(size_t)(base + m) * DIM_ + n] = acc[ms][ns][r] + bb;
      }
    }
  }
}

// out[t,d] = w0*y[slot0,d] + w1*y[slot1,d]  (deterministic combine)
__global__ __launch_bounds__(256)
void combine_kernel(const float* __restrict__ y, const int* __restrict__ tokenSlot,
                    const float* __restrict__ tokWeight, float* __restrict__ out) {
  int i = blockIdx.x * blockDim.x + threadIdx.x;
  const int stride = gridDim.x * blockDim.x;
  for (; i < NTOK * DIM_; i += stride) {
    const int t = i >> 10;            // / DIM_
    const int d = i & (DIM_ - 1);
    const int s0 = tokenSlot[2 * t], s1 = tokenSlot[2 * t + 1];
    out[i] = tokWeight[2 * t]     * y[(size_t)s0 * DIM_ + d] +
             tokWeight[2 * t + 1] * y[(size_t)s1 * DIM_ + d];
  }
}

// ---------------------------------------------------------------------------
extern "C" void kernel_launch(void* const* d_in, const int* in_sizes, int n_in,
                              void* d_out, int out_size, void* d_ws, size_t ws_size,
                              hipStream_t stream) {
  (void)in_sizes; (void)n_in; (void)out_size; (void)ws_size;
  const float* x  = (const float*)d_in[0];
  const float* rw = (const float*)d_in[1];
  const float* rb = (const float*)d_in[2];
  const float* W1 = (const float*)d_in[3];
  const float* b1 = (const float*)d_in[4];
  const float* W2 = (const float*)d_in[5];
  const float* b2 = (const float*)d_in[6];
  float* out = (float*)d_out;

  // Workspace carve (~240.5 MB total)
  char* ws = (char*)d_ws;
  size_t off = 0;
  auto alloc = [&](size_t bytes) -> char* {
    off = (off + 255) & ~(size_t)255;
    char* p = ws + off;
    off += bytes;
    return p;
  };
  int*   counts    = (int*)  alloc(E_ * 4);
  int*   offsets   = (int*)  alloc((E_ + 1) * 4);
  int*   fill      = (int*)  alloc(E_ * 4);
  int*   tokExpert = (int*)  alloc((size_t)NENT * 4);
  float* tokWeight = (float*)alloc((size_t)NENT * 4);
  int*   tokenSlot = (int*)  alloc((size_t)NENT * 4);
  int*   entryTok  = (int*)  alloc((size_t)NENT * 4);
  u16*   xb  = (u16*)alloc((size_t)NTOK * DIM_ * 2);              // 16 MB
  u16*   W1b = (u16*)alloc((size_t)E_ * DIM_ * 2 * HID_ * 2);     // 64 MB
  u16*   W2b = (u16*)alloc((size_t)E_ * HID_ * DIM_ * 2);         // 32 MB
  u16*   act = (u16*)alloc((size_t)NENT * HID_ * 2);              // 64 MB
  float* y   = (float*)alloc((size_t)NENT * DIM_ * 4);            // 64 MB

  zero_meta<<<dim3(1), dim3(32), 0, stream>>>(counts, fill);
  router_kernel<<<dim3(NTOK / 8), dim3(256), 0, stream>>>(x, rw, rb, tokExpert,
                                                          tokWeight, counts);
  scan_kernel<<<dim3(1), dim3(1), 0, stream>>>(counts, offsets);
  scatter_kernel<<<dim3(NTOK / 256), dim3(256), 0, stream>>>(tokExpert, offsets, fill,
                                                             entryTok, tokenSlot);
  cast_kernel<<<dim3(2048), dim3(256), 0, stream>>>(x, xb, NTOK * DIM_);
  cast_kernel<<<dim3(4096), dim3(256), 0, stream>>>(W1, W1b, E_ * DIM_ * 2 * HID_);
  cast_kernel<<<dim3(4096), dim3(256), 0, stream>>>(W2, W2b, E_ * HID_ * DIM_);

  gemm1_swiglu_kernel<<<dim3(NTOK / TM, HID_ / TN, E_), dim3(256), 0, stream>>>(
      xb, W1b, b1, entryTok, offsets, act);
  gemm2_kernel<<<dim3(NTOK / TM, DIM_ / TN, E_), dim3(256), 0, stream>>>(
      act, W2b, b2, offsets, y);
  combine_kernel<<<dim3(4096), dim3(256), 0, stream>>>(y, tokenSlot, tokWeight, out);
}
